// MoEMLP_4063039062645
// MI455X (gfx1250) — matrix operand order repaired and checked by reading, not compiled, not run on hardware
//
#include <hip/hip_runtime.h>
#include <hip/hip_bf16.h>

typedef __bf16 bf16;
typedef __attribute__((ext_vector_type(16))) __bf16 bf16x16;
typedef __attribute__((ext_vector_type(8)))  __bf16 bf16x8;
typedef __attribute__((ext_vector_type(4)))  __bf16 bf16x4;
typedef __attribute__((ext_vector_type(2)))  __bf16 bf16x2;
typedef __attribute__((ext_vector_type(8)))  float  f32x8;

#define N_TOK 4096
#define HDIM  1024
#define DDIM  2048
#define NEXP  8
#define TOPK  2
#define TTILE 32      // tokens per workgroup tile
#define KC    32      // K chunk per stage
#define GCOLS 512     // output columns per group (8 waves * 4 tiles * 16)
#define KPAD  32      // K stride in stage buffer (16B-aligned rows)

// dynamic LDS layout (bytes)
#define OFF_X     0                          // 32*1024 bf16 = 65536
#define OFF_HID   65536                      // 32*2048 bf16 = 131072
#define OFF_STAGE (65536 + 131072)           // 512*32 bf16 = 32768
#define OFF_TOK   (OFF_STAGE + 32768)        // 32 int
#define OFF_WT    (OFF_TOK + 128)            // 32 float
#define LDS_BYTES (OFF_WT + 128)             // 229632 (< 320KB/WGP)

__global__ void moe_zero_counts(int* counts) {
    if (threadIdx.x < NEXP) counts[threadIdx.x] = 0;
}

__global__ void moe_route(const int* __restrict__ idx, const float* __restrict__ ew,
                          const int* __restrict__ mask, int* __restrict__ counts,
                          int* __restrict__ ids, float* __restrict__ wv) {
    int j = blockIdx.x * blockDim.x + threadIdx.x;
    if (j >= N_TOK * TOPK) return;
    int n = j >> 1;
    int e = idx[j];
    float w = ew[j] * (float)mask[n];
    int pos = atomicAdd(&counts[e], 1);
    ids[e * N_TOK + pos] = n;
    wv[e * N_TOK + pos] = w;
}

__global__ void moe_residual(const float* __restrict__ x, float* __restrict__ out) {
    int i = blockIdx.x * blockDim.x + threadIdx.x;   // one float4 each
    ((float4*)out)[i] = ((const float4*)x)[i];
}

// Stage a KC x GCOLS fp32 chunk of W (row stride `stride`) into Wst[c][k] as bf16,
// transposed, packing row-pairs into b32 stores.
__device__ __forceinline__ void stage_w(const float* __restrict__ Wsrc, int stride,
                                        int kc, int gbase, bf16* __restrict__ Wst, int tid) {
    #pragma unroll 4
    for (int i = 0; i < 8; ++i) {
        int u  = i * 256 + tid;               // 16 row-pairs * 128 float4-cols
        int pr = u >> 7;                      // row pair 0..15
        int c4 = u & 127;                     // float4 column 0..127
        const float* p0 = Wsrc + (size_t)(kc * KC + 2 * pr)     * stride + gbase + c4 * 4;
        const float* p1 = Wsrc + (size_t)(kc * KC + 2 * pr + 1) * stride + gbase + c4 * 4;
        float4 va = *(const float4*)p0;
        float4 vb = *(const float4*)p1;
        const float* fa = (const float*)&va;
        const float* fb = (const float*)&vb;
        #pragma unroll
        for (int j = 0; j < 4; ++j) {
            bf16x2 p; p[0] = (bf16)fa[j]; p[1] = (bf16)fb[j];
            *(bf16x2*)(Wst + (c4 * 4 + j) * KPAD + 2 * pr) = p;
        }
    }
}

__device__ __forceinline__ bf16x16 load_afrag(const bf16* __restrict__ base) {
    // elements 0..7 = K k0..k0+7 ; 8..15 = K k0+16..k0+23 (ISA 16-bit A layout)
    bf16x8 lo = *(const bf16x8*)base;
    bf16x8 hi = *(const bf16x8*)(base + 16);
    bf16x16 a;
    #pragma unroll
    for (int q = 0; q < 8; ++q) { a[q] = lo[q]; a[8 + q] = hi[q]; }
    return a;
}

__launch_bounds__(256)
__global__ void moe_wmma(const float* __restrict__ x,
                         const float* __restrict__ down,   // [E][H][D]
                         const float* __restrict__ up,     // [E][D][H]
                         const int*   __restrict__ counts,
                         const int*   __restrict__ ids,
                         const float* __restrict__ wv,
                         float* __restrict__ out) {
    extern __shared__ char lds[];
    bf16*  Xs  = (bf16*)(lds + OFF_X);      // [32][1024]
    bf16*  Hid = (bf16*)(lds + OFF_HID);    // [32][2048]
    bf16*  Wst = (bf16*)(lds + OFF_STAGE);  // [512][KPAD]
    int*   tok = (int*)(lds + OFF_TOK);
    float* twt = (float*)(lds + OFF_WT);

    const int e     = blockIdx.y;
    const int tile0 = blockIdx.x * TTILE;
    const int cnt   = counts[e];
    if (tile0 >= cnt) return;                 // uniform per-block exit

    const int tid     = threadIdx.x;
    const int lane    = tid & 31;
    const int wave    = tid >> 5;             // 0..7
    const int halfsel = (lane >> 4);          // 0|1
    const int l16     = lane & 15;

    if (tid < TTILE) {
        int slot = tile0 + tid;
        if (slot < cnt) { tok[tid] = ids[e * N_TOK + slot]; twt[tid] = wv[e * N_TOK + slot]; }
        else            { tok[tid] = 0;                      twt[tid] = 0.0f; }
    }
    __syncthreads();

    // ---- stage X tile (fp32 -> bf16) into LDS: 32 x 1024 ----
    #pragma unroll 4
    for (int i = 0; i < 32; ++i) {
        int fid = i * 256 + tid;              // 8192 float4 total
        int row = fid >> 6;                   // 64 float4 per row
        int c4  = fid & 63;
        float4 v = ((const float4*)(x + (size_t)tok[row] * HDIM))[c4];
        bf16x4 b; b[0] = (bf16)v.x; b[1] = (bf16)v.y; b[2] = (bf16)v.z; b[3] = (bf16)v.w;
        *(bf16x4*)(Xs + row * HDIM + c4 * 4) = b;
    }
    __syncthreads();

    // =========== Pass A: Hid = relu(X @ Wd[e]) , K = H = 1024 ===========
    const float* Wd = down + (size_t)e * HDIM * DDIM;
    for (int g = 0; g < DDIM / GCOLS; ++g) {            // 4 column groups of 512
        f32x8 acc[2][4] = {};
        for (int kc = 0; kc < HDIM / KC; ++kc) {        // 32 K-chunks
            if (kc + 1 < HDIM / KC) {                   // uniform: prefetch next chunk
                const char* pf = (const char*)(Wd + (size_t)((kc + 1) * KC + (tid >> 3)) * DDIM
                                               + g * GCOLS + (tid & 7) * 64);
                __builtin_prefetch(pf, 0, 1);
            }
            stage_w(Wd, DDIM, kc, g * GCOLS, Wst, tid);
            __syncthreads();
            bf16x16 a0 = load_afrag(Xs + (l16)      * HDIM + kc * KC + halfsel * 8);
            bf16x16 a1 = load_afrag(Xs + (16 + l16) * HDIM + kc * KC + halfsel * 8);
            #pragma unroll
            for (int t = 0; t < 4; ++t) {
                const bf16* bb = Wst + (wave * 64 + t * 16 + l16) * KPAD + halfsel * 8;
                bf16x8 b0 = *(const bf16x8*)bb;
                bf16x8 b1 = *(const bf16x8*)(bb + 16);
                bf16x16 b;
                #pragma unroll
                for (int q = 0; q < 8; ++q) { b[q] = b0[q]; b[8 + q] = b1[q]; }
                acc[0][t] = __builtin_amdgcn_wmma_f32_16x16x32_bf16(
                    false, a0, false, b, (short)0, acc[0][t], false, false);
                acc[1][t] = __builtin_amdgcn_wmma_f32_16x16x32_bf16(
                    false, a1, false, b, (short)0, acc[1][t], false, false);
            }
            __syncthreads();
        }
        // relu + write bf16 into Hid
        #pragma unroll
        for (int m = 0; m < 2; ++m) {
            #pragma unroll
            for (int t = 0; t < 4; ++t) {
                int n0 = g * GCOLS + wave * 64 + t * 16;
                #pragma unroll
                for (int r = 0; r < 8; ++r) {
                    float v = acc[m][t][r]; v = v > 0.0f ? v : 0.0f;
                    int M = m * 16 + r + halfsel * 8;
                    Hid[M * DDIM + n0 + l16] = (bf16)v;
                }
            }
        }
    }
    __syncthreads();

    // =========== Pass B: y = Hid @ Wu[e], scale, atomic add ; K = D = 2048 ===========
    const float* Wu = up + (size_t)e * DDIM * HDIM;
    for (int g = 0; g < HDIM / GCOLS; ++g) {            // 2 column groups of 512
        f32x8 acc[2][4] = {};
        for (int kc = 0; kc < DDIM / KC; ++kc) {        // 64 K-chunks
            if (kc + 1 < DDIM / KC) {                   // uniform: prefetch next chunk
                const char* pf = (const char*)(Wu + (size_t)((kc + 1) * KC + (tid >> 3)) * HDIM
                                               + g * GCOLS + (tid & 7) * 64);
                __builtin_prefetch(pf, 0, 1);
            }
            stage_w(Wu, HDIM, kc, g * GCOLS, Wst, tid);
            __syncthreads();
            bf16x16 a0 = load_afrag(Hid + (l16)      * DDIM + kc * KC + halfsel * 8);
            bf16x16 a1 = load_afrag(Hid + (16 + l16) * DDIM + kc * KC + halfsel * 8);
            #pragma unroll
            for (int t = 0; t < 4; ++t) {
                const bf16* bb = Wst + (wave * 64 + t * 16 + l16) * KPAD + halfsel * 8;
                bf16x8 b0 = *(const bf16x8*)bb;
                bf16x8 b1 = *(const bf16x8*)(bb + 16);
                bf16x16 b;
                #pragma unroll
                for (int q = 0; q < 8; ++q) { b[q] = b0[q]; b[8 + q] = b1[q]; }
                acc[0][t] = __builtin_amdgcn_wmma_f32_16x16x32_bf16(
                    false, a0, false, b, (short)0, acc[0][t], false, false);
                acc[1][t] = __builtin_amdgcn_wmma_f32_16x16x32_bf16(
                    false, a1, false, b, (short)0, acc[1][t], false, false);
            }
            __syncthreads();
        }
        // scale by routing weight, atomic-add into out
        #pragma unroll
        for (int m = 0; m < 2; ++m) {
            #pragma unroll
            for (int t = 0; t < 4; ++t) {
                int col = g * GCOLS + wave * 64 + t * 16 + l16;
                #pragma unroll
                for (int r = 0; r < 8; ++r) {
                    int M = m * 16 + r + halfsel * 8;
                    float v = acc[m][t][r] * twt[M];
                    atomicAdd(out + (size_t)tok[M] * HDIM + col, v);
                }
            }
        }
    }
}

extern "C" void kernel_launch(void* const* d_in, const int* in_sizes, int n_in,
                              void* d_out, int out_size, void* d_ws, size_t ws_size,
                              hipStream_t stream) {
    const float* x    = (const float*)d_in[0];
    const int*   mask = (const int*)d_in[1];
    const float* ew   = (const float*)d_in[2];
    const int*   idx  = (const int*)d_in[3];
    const float* down = (const float*)d_in[4];
    const float* up   = (const float*)d_in[5];
    float* out = (float*)d_out;

    // workspace: counts[8] | ids[8*4096] int | wv[8*4096] float
    int*   counts = (int*)d_ws;
    int*   ids    = counts + NEXP;
    float* wvv    = (float*)(ids + NEXP * N_TOK);

    moe_zero_counts<<<1, 32, 0, stream>>>(counts);
    moe_route<<<(N_TOK * TOPK + 255) / 256, 256, 0, stream>>>(idx, ew, mask, counts, ids, wvv);
    moe_residual<<<(N_TOK * HDIM / 4) / 256, 256, 0, stream>>>(x, out);

    dim3 grid(N_TOK / TTILE, NEXP);
    moe_wmma<<<grid, 256, LDS_BYTES, stream>>>(x, down, up, counts, ids, wvv, out);
    (void)in_sizes; (void)n_in; (void)out_size; (void)ws_size;
}
